// TransformerBlock_893353197782
// MI455X (gfx1250) — matrix operand order repaired and checked by reading, not compiled, not run on hardware
//
#include <hip/hip_runtime.h>
#include <hip/hip_bf16.h>
#include <math.h>

typedef __attribute__((ext_vector_type(16))) __bf16 bf16x16;
typedef __attribute__((ext_vector_type(8)))  float  f32x8;

#define E_DIM 128
#define H_HEADS 8
#define HS 16
#define HID 65536
#define BB 2
#define TT 2048
#define NTOK (BB*TT)
#define FFN_SEG 8               // split-K segments over HID
#define NCH_TOTAL (HID/32)      // 2048 chunks of 32 hidden cols
#define NCH_SEG (NCH_TOTAL/FFN_SEG)

__device__ __forceinline__ __bf16 to_bf16(float f) { return (__bf16)f; }

__device__ __forceinline__ f32x8 zero8() {
  f32x8 z;
#pragma unroll
  for (int i = 0; i < 8; ++i) z[i] = 0.0f;
  return z;
}

__device__ __forceinline__ f32x8 wmma_bf16(bf16x16 a, bf16x16 b, f32x8 c) {
  return __builtin_amdgcn_wmma_f32_16x16x32_bf16(false, a, false, b, (short)0, c, false, false);
}

// ---- CDNA5 async global->LDS copy (ASYNCcnt path), guarded by builtin probe ----
#if defined(__AMDGCN__) && __has_builtin(__builtin_amdgcn_global_load_async_to_lds_b128)
#define HAVE_ASYNC_LDS 1
typedef int v4i __attribute__((vector_size(4 * sizeof(int))));
typedef __attribute__((address_space(1))) v4i as1_v4i;
typedef __attribute__((address_space(3))) v4i as3_v4i;
__device__ __forceinline__ void async_copy16(const __bf16* g, __bf16* l) {
  __builtin_amdgcn_global_load_async_to_lds_b128((as1_v4i*)(void*)g, (as3_v4i*)(void*)l, 0, 0);
}
__device__ __forceinline__ void wait_async0() {
#if __has_builtin(__builtin_amdgcn_s_wait_asynccnt)
  __builtin_amdgcn_s_wait_asynccnt(0);
#else
  asm volatile("s_wait_asynccnt 0x0" ::: "memory");
#endif
}
#else
#define HAVE_ASYNC_LDS 0
#endif

// A-fragment slot <-> k mapping (16x32 bf16 A):
//   slot i holds k = hf*8 + (i&7) + ((i>>3)<<4)  (hf = lane>>4)
// inverse: for value at column k: lane_add = 16*((k>>3)&1), slot = (k&7)|((k>>4)<<3)
__device__ __forceinline__ int a_slot(int k)     { return (k & 7) | ((k >> 4) << 3); }
__device__ __forceinline__ int a_lane_off(int k) { return ((k >> 3) & 1) << 4; }

// load 16x32 A fragment from row-major bf16 matrix (lda in elements)
__device__ __forceinline__ bf16x16 load_a_rm(const __bf16* src, int lda, int lane) {
  const int m = lane & 15, hf = lane >> 4;
  bf16x16 a;
#pragma unroll
  for (int i = 0; i < 16; i += 2) {
    int k = hf*8 + (i & 7) + ((i >> 3) << 4);
    a[i]   = src[m*lda + k];
    a[i+1] = src[m*lda + k + 1];
  }
  return a;
}

// -------- weight packing: fp32 row-major [K x N] -> bf16 B-fragment tiles --------
// tile = 32x16; within tile: lane-major, 16 elems/lane; lane=(n,hf), slot i -> k=hf*8+(i&7)+((i>>3)<<4)
__global__ void pack_b_kernel(const float* __restrict__ src, __bf16* __restrict__ dst,
                              int K, int N, int col_order) {
  const size_t batch = blockIdx.y;
  src += batch * (size_t)K * N;
  dst += batch * (size_t)K * N;
  size_t idx = (size_t)blockIdx.x * blockDim.x + threadIdx.x;
  size_t total = (size_t)K * N;
  if (idx >= total) return;
  size_t tile = idx >> 9;
  int r = (int)(idx & 511);
  int lane = r >> 4, i = r & 15;
  int ktiles = K >> 5, ntiles = N >> 4;
  int kt, nt;
  if (col_order) { nt = (int)(tile / ktiles); kt = (int)(tile % ktiles); }
  else           { kt = (int)(tile / ntiles); nt = (int)(tile % ntiles); }
  int n = lane & 15, hf = lane >> 4;
  int k   = kt*32 + hf*8 + (i & 7) + ((i >> 3) << 4);
  int col = nt*16 + n;
  dst[idx] = to_bf16(src[(size_t)k*N + col]);
}

// -------- LayerNorm: one block per row of 128 --------
__global__ __launch_bounds__(128) void ln_kernel(const float* __restrict__ x, const float* __restrict__ g,
                                                 const float* __restrict__ be, __bf16* __restrict__ out) {
  __shared__ float red[128];
  const int row = blockIdx.x, tid = threadIdx.x;
  float v = x[(size_t)row*E_DIM + tid];
  red[tid] = v;
  __syncthreads();
#pragma unroll
  for (int s = 64; s > 0; s >>= 1) { if (tid < s) red[tid] += red[tid + s]; __syncthreads(); }
  float mean = red[0] * (1.0f/E_DIM);
  __syncthreads();
  float d = v - mean;
  red[tid] = d*d;
  __syncthreads();
#pragma unroll
  for (int s = 64; s > 0; s >>= 1) { if (tid < s) red[tid] += red[tid + s]; __syncthreads(); }
  float var = red[0] * (1.0f/E_DIM);
  float rr = rsqrtf(var + 1e-5f);
  out[(size_t)row*E_DIM + tid] = to_bf16(d * rr * g[tid] + be[tid]);
}

// -------- QKV projection: one wave per (b,h,t-tile16) --------
__global__ __launch_bounds__(256) void qkv_kernel(const __bf16* __restrict__ hb, const __bf16* __restrict__ Wqp,
                                                  const __bf16* __restrict__ Wkp, const __bf16* __restrict__ Wvp,
                                                  __bf16* __restrict__ qb, __bf16* __restrict__ ktp,
                                                  __bf16* __restrict__ vtp) {
  const int tid = threadIdx.x, lane = tid & 31, nc = lane & 15, hf = lane >> 4;
  const int wave = blockIdx.x * 8 + (tid >> 5);
  const int bh = wave >> 7, tt = wave & 127;
  const int b = bh >> 3, h = bh & 7;
  const int t0 = tt * 16;
  const size_t arow = (size_t)b * TT + t0;

  bf16x16 af[4];
#pragma unroll
  for (int kk = 0; kk < 4; ++kk) af[kk] = load_a_rm(hb + arow*E_DIM + kk*32, E_DIM, lane);

  { // Q
    f32x8 c = zero8();
#pragma unroll
    for (int kk = 0; kk < 4; ++kk) {
      bf16x16 bw = *(const bf16x16*)(Wqp + (size_t)h*2048 + kk*512 + lane*16);
      c = wmma_bf16(af[kk], bw, c);
    }
#pragma unroll
    for (int v = 0; v < 8; ++v)
      qb[((size_t)bh*TT + t0 + v + 8*hf)*HS + nc] = to_bf16(c[v]);
  }
  { // K (transposed B tiles, K-dim padded 16->32 with zeros)
    f32x8 c = zero8();
#pragma unroll
    for (int kk = 0; kk < 4; ++kk) {
      bf16x16 bw = *(const bf16x16*)(Wkp + (size_t)h*2048 + kk*512 + lane*16);
      c = wmma_bf16(af[kk], bw, c);
    }
    __bf16* tile = ktp + ((size_t)bh*(TT/16) + tt)*512;
#pragma unroll
    for (int v = 0; v < 8; ++v) {
      tile[(v + 8*hf + a_lane_off(nc))*16 + a_slot(nc)] = to_bf16(c[v]);   // slots 0..7: k = d (table order)
      tile[lane*16 + 8 + v] = to_bf16(0.f);    // slots 8..15: zero pad
    }
  }
  { // V (B-packed 32-row tiles)
    f32x8 c = zero8();
#pragma unroll
    for (int kk = 0; kk < 4; ++kk) {
      bf16x16 bw = *(const bf16x16*)(Wvp + (size_t)h*2048 + kk*512 + lane*16);
      c = wmma_bf16(af[kk], bw, c);
    }
    __bf16* tile = vtp + ((size_t)bh*(TT/32) + (tt >> 1))*512;
    const int lb = nc + 16*hf;
#pragma unroll
    for (int v = 0; v < 8; ++v)
      tile[lb*16 + (v + 8*(tt & 1))] = to_bf16(c[v]);
  }
}

// -------- causal flash attention: one wave per (b,h,t-tile16) --------
__global__ __launch_bounds__(256) void flash_kernel(const __bf16* __restrict__ qb, const __bf16* __restrict__ ktp,
                                                    const __bf16* __restrict__ vtp, __bf16* __restrict__ ob) {
  __shared__ __attribute__((aligned(32))) __bf16 psm[8*512];
  const int tid = threadIdx.x, lane = tid & 31, nc = lane & 15, hf = lane >> 4;
  const int wave = blockIdx.x * 8 + (tid >> 5);
  const int bh = wave >> 7, tt = wave & 127;
  const int t0 = tt * 16;
  __bf16* p = psm + (tid >> 5) * 512;

  const __bf16* qs = qb + ((size_t)bh * TT + t0) * HS;
  bf16x16 qf;
#pragma unroll
  for (int v = 0; v < 4; ++v) {
    int k = hf*8 + 2*v;
    qf[2*v]   = qs[nc*HS + k];
    qf[2*v+1] = qs[nc*HS + k + 1];
  }
#pragma unroll
  for (int v = 4; v < 8; ++v) { qf[2*v] = to_bf16(0.f); qf[2*v+1] = to_bf16(0.f); }

  const float scale = 0.088388347648318447f; // E^-0.5
  float rm[8], rs[8];
#pragma unroll
  for (int v = 0; v < 8; ++v) { rm[v] = -INFINITY; rs[v] = 0.f; }
  f32x8 acc = zero8();

  const int thi = t0 + 15;
  for (int s0 = 0; s0 <= thi; s0 += 32) {
    f32x8 sc0, sc1;
    {
      bf16x16 kf = *(const bf16x16*)(ktp + ((size_t)bh*(TT/16) + (s0 >> 4))*512 + lane*16);
      sc0 = wmma_bf16(qf, kf, zero8());
    }
    if (s0 + 16 <= thi) {
      bf16x16 kf = *(const bf16x16*)(ktp + ((size_t)bh*(TT/16) + (s0 >> 4) + 1)*512 + lane*16);
      sc1 = wmma_bf16(qf, kf, zero8());
    } else {
#pragma unroll
      for (int v = 0; v < 8; ++v) sc1[v] = -INFINITY;
    }
#pragma unroll
    for (int v = 0; v < 8; ++v) {
      int row = t0 + v + 8*hf;
      sc0[v] = ((s0 + nc)      <= row) ? sc0[v]*scale : -INFINITY;
      sc1[v] = ((s0 + 16 + nc) <= row) ? sc1[v]*scale : -INFINITY;
    }
    const int ii0 = a_slot(nc),      lo0 = a_lane_off(nc);
    const int ii1 = a_slot(16 + nc), lo1 = a_lane_off(16 + nc);
#pragma unroll
    for (int v = 0; v < 8; ++v) {
      float cm = fmaxf(sc0[v], sc1[v]);
#pragma unroll
      for (int o = 8; o >= 1; o >>= 1) cm = fmaxf(cm, __shfl_xor(cm, o, 32));
      float nm = fmaxf(rm[v], cm);
      float so = __expf(rm[v] - nm);
      float p0 = __expf(sc0[v] - nm);
      float p1 = __expf(sc1[v] - nm);
      int mrow = v + 8*hf;
      p[(mrow + lo0)*16 + ii0] = to_bf16(p0);
      p[(mrow + lo1)*16 + ii1] = to_bf16(p1);
      float ps = p0 + p1;
#pragma unroll
      for (int o = 8; o >= 1; o >>= 1) ps += __shfl_xor(ps, o, 32);
      rs[v] = rs[v]*so + ps;
      rm[v] = nm;
      acc[v] *= so;
    }
    bf16x16 pf = *(const bf16x16*)(p + lane*16);
    bf16x16 vf = *(const bf16x16*)(vtp + ((size_t)bh*(TT/32) + (s0 >> 5))*512 + lane*16);
    acc = wmma_bf16(pf, vf, acc);
  }
  const int b = bh >> 3, h = bh & 7;
#pragma unroll
  for (int v = 0; v < 8; ++v) {
    float o = acc[v] / rs[v];
    int row = t0 + v + 8*hf;
    ob[((size_t)b*TT + row)*E_DIM + h*HS + nc] = to_bf16(o);
  }
}

// -------- output projection + residual --------
__global__ __launch_bounds__(256) void proj_kernel(const __bf16* __restrict__ ob, const __bf16* __restrict__ Wop,
                                                   const float* __restrict__ bo, const float* __restrict__ x,
                                                   float* __restrict__ x1) {
  const int tid = threadIdx.x, lane = tid & 31, nc = lane & 15, hf = lane >> 4;
  const int wave = blockIdx.x * 8 + (tid >> 5);
  const int ttile = wave >> 3, nt = wave & 7;
  const int r0 = ttile * 16;
  bf16x16 af[4];
#pragma unroll
  for (int kk = 0; kk < 4; ++kk) af[kk] = load_a_rm(ob + (size_t)r0*E_DIM + kk*32, E_DIM, lane);
  f32x8 c = zero8();
#pragma unroll
  for (int kk = 0; kk < 4; ++kk) {
    bf16x16 bw = *(const bf16x16*)(Wop + (size_t)(nt*4 + kk)*512 + lane*16);
    c = wmma_bf16(af[kk], bw, c);
  }
#pragma unroll
  for (int v = 0; v < 8; ++v) {
    int row = r0 + v + 8*hf, col = nt*16 + nc;
    x1[(size_t)row*E_DIM + col] = x[(size_t)row*E_DIM + col] + c[v] + bo[col];
  }
}

// -------- fused FFN, split-K over FFN_SEG segments --------
// grid = FFN_SEG * 32 blocks; block = 8 waves = 8 token tiles; double-buffered LDS weight chunks
__global__ __launch_bounds__(256) void ffn_kernel(const __bf16* __restrict__ h2, const __bf16* __restrict__ W1p,
                                                  const __bf16* __restrict__ W2p, const float* __restrict__ bf1,
                                                  float* __restrict__ part) {
  __shared__ __attribute__((aligned(32))) __bf16 smem[20480];
  const int tid = threadIdx.x, lane = tid & 31, nc = lane & 15, hf = lane >> 4;
  const int wv = tid >> 5;
  const int tgroup = blockIdx.x & 31;
  const int seg = blockIdx.x >> 5;
  const int r0 = (tgroup * 8 + wv) * 16;
  const int jc0 = seg * NCH_SEG;
  __bf16* pbuf = smem + 16384 + wv*512;

  bf16x16 af[4];
#pragma unroll
  for (int kk = 0; kk < 4; ++kk) af[kk] = load_a_rm(h2 + (size_t)r0*E_DIM + kk*32, E_DIM, lane);

  f32x8 acc[8];
#pragma unroll
  for (int nt = 0; nt < 8; ++nt) acc[nt] = zero8();

  // stage one 16 KB chunk (W1 8KB + W2 8KB) into buffer (jc&1)
  auto stage = [&](int jc) {
    const __bf16* s1 = W1p + (size_t)jc*4096;
    const __bf16* s2 = W2p + (size_t)jc*4096;
    __bf16* d = smem + (jc & 1) * 8192;
#if HAVE_ASYNC_LDS
#pragma unroll
    for (int i = 0; i < 2; ++i) {
      int e = (tid + 256*i) * 8;          // 8 bf16 = 16 bytes per transfer
      async_copy16(s1 + e, d + e);
      async_copy16(s2 + e, d + 4096 + e);
    }
#else
    const uint4* s1v = (const uint4*)s1;
    const uint4* s2v = (const uint4*)s2;
    uint4* d1 = (uint4*)d;
    uint4* d2 = d1 + 512;
#pragma unroll
    for (int i = 0; i < 2; ++i) { d1[tid + 256*i] = s1v[tid + 256*i]; d2[tid + 256*i] = s2v[tid + 256*i]; }
#endif
  };

  stage(jc0);
#if HAVE_ASYNC_LDS
  wait_async0();
#endif
  __syncthreads();

  for (int jj = 0; jj < NCH_SEG; ++jj) {
    const int jc = jc0 + jj;
    __bf16* cur = smem + (jc & 1) * 8192;
    if (jj + 1 < NCH_SEG) stage(jc + 1);   // overlaps with compute below

    const int j0 = jc * 32;
    // hidden tile [16 x 32] = relu(h2 @ W1 + b1); scatter into A-fragment layout
#pragma unroll
    for (int sub = 0; sub < 2; ++sub) {
      f32x8 hC = zero8();
#pragma unroll
      for (int kk = 0; kk < 4; ++kk) {
        bf16x16 bw = *(const bf16x16*)(cur + (sub*4 + kk)*512 + lane*16);
        hC = wmma_bf16(af[kk], bw, hC);
      }
      float bias = bf1[j0 + sub*16 + nc];
      const int kl = sub*16 + nc;
      const int ii = a_slot(kl), lo = a_lane_off(kl);
#pragma unroll
      for (int v = 0; v < 8; ++v) {
        float hval = fmaxf(hC[v] + bias, 0.0f);
        pbuf[(v + 8*hf + lo)*16 + ii] = to_bf16(hval);
      }
    }
    bf16x16 pf = *(const bf16x16*)(pbuf + lane*16);
    const __bf16* w2c = cur + 4096;
#pragma unroll
    for (int nt = 0; nt < 8; ++nt) {
      bf16x16 bw = *(const bf16x16*)(w2c + nt*512 + lane*16);
      acc[nt] = wmma_bf16(pf, bw, acc[nt]);
    }
#if HAVE_ASYNC_LDS
    wait_async0();
#endif
    __syncthreads();
  }

  float* pp = part + (size_t)seg * NTOK * E_DIM;
#pragma unroll
  for (int nt = 0; nt < 8; ++nt) {
#pragma unroll
    for (int v = 0; v < 8; ++v) {
      int row = r0 + v + 8*hf, col = nt*16 + nc;
      pp[(size_t)row*E_DIM + col] = acc[nt][v];
    }
  }
}

// -------- FFN split-K reduction + residual + bias --------
__global__ __launch_bounds__(256) void ffn_reduce_kernel(const float* __restrict__ part,
                                                         const float* __restrict__ x1,
                                                         const float* __restrict__ bf2,
                                                         float* __restrict__ out) {
  size_t idx = (size_t)blockIdx.x * 256 + threadIdx.x;
  int col = (int)(idx & (E_DIM - 1));
  float s = x1[idx] + bf2[col];
#pragma unroll
  for (int g = 0; g < FFN_SEG; ++g) s += part[(size_t)g * NTOK * E_DIM + idx];
  out[idx] = s;
}

extern "C" void kernel_launch(void* const* d_in, const int* in_sizes, int n_in,
                              void* d_out, int out_size, void* d_ws, size_t ws_size,
                              hipStream_t stream) {
  const float* x   = (const float*)d_in[0];
  const float* Wk  = (const float*)d_in[1];
  const float* Wq  = (const float*)d_in[2];
  const float* Wv  = (const float*)d_in[3];
  const float* Wo  = (const float*)d_in[4];
  const float* bo  = (const float*)d_in[5];
  const float* g1  = (const float*)d_in[6];
  const float* be1 = (const float*)d_in[7];
  const float* g2  = (const float*)d_in[8];
  const float* be2 = (const float*)d_in[9];
  const float* W1  = (const float*)d_in[10];
  const float* bf1 = (const float*)d_in[11];
  const float* W2  = (const float*)d_in[12];
  const float* bf2 = (const float*)d_in[13];
  float* out = (float*)d_out;

  char* w = (char*)d_ws;
  size_t off = 0;
  auto alloc = [&](size_t bytes) -> char* {
    char* p = w + off;
    off += (bytes + 255) & ~(size_t)255;
    return p;
  };
  __bf16* hb   = (__bf16*)alloc((size_t)NTOK*E_DIM*2);
  __bf16* qb   = (__bf16*)alloc((size_t)NTOK*E_DIM*2);
  __bf16* ktp  = (__bf16*)alloc((size_t)16*(TT/16)*512*2);
  __bf16* vtp  = (__bf16*)alloc((size_t)16*(TT/32)*512*2);
  __bf16* ob   = (__bf16*)alloc((size_t)NTOK*E_DIM*2);
  float*  x1   = (float*) alloc((size_t)NTOK*E_DIM*4);
  __bf16* h2b  = (__bf16*)alloc((size_t)NTOK*E_DIM*2);
  float*  part = (float*) alloc((size_t)FFN_SEG*NTOK*E_DIM*4);
  __bf16* Wqp  = (__bf16*)alloc((size_t)H_HEADS*E_DIM*HS*2);
  __bf16* Wkp  = (__bf16*)alloc((size_t)H_HEADS*E_DIM*HS*2);
  __bf16* Wvp  = (__bf16*)alloc((size_t)H_HEADS*E_DIM*HS*2);
  __bf16* Wop  = (__bf16*)alloc((size_t)E_DIM*E_DIM*2);
  __bf16* W1p  = (__bf16*)alloc((size_t)E_DIM*HID*2);
  __bf16* W2p  = (__bf16*)alloc((size_t)HID*E_DIM*2);

  pack_b_kernel<<<dim3(8, H_HEADS), 256, 0, stream>>>(Wq, Wqp, E_DIM, HS, 1);
  pack_b_kernel<<<dim3(8, H_HEADS), 256, 0, stream>>>(Wk, Wkp, E_DIM, HS, 1);
  pack_b_kernel<<<dim3(8, H_HEADS), 256, 0, stream>>>(Wv, Wvp, E_DIM, HS, 1);
  pack_b_kernel<<<dim3(64, 1), 256, 0, stream>>>(Wo, Wop, E_DIM, E_DIM, 1);
  pack_b_kernel<<<dim3(32768, 1), 256, 0, stream>>>(W1, W1p, E_DIM, HID, 1);   // column-of-tiles major
  pack_b_kernel<<<dim3(32768, 1), 256, 0, stream>>>(W2, W2p, HID, E_DIM, 0);   // row-of-tiles major

  ln_kernel<<<NTOK, 128, 0, stream>>>(x, g1, be1, hb);
  qkv_kernel<<<256, 256, 0, stream>>>(hb, Wqp, Wkp, Wvp, qb, ktp, vtp);
  flash_kernel<<<256, 256, 0, stream>>>(qb, ktp, vtp, ob);
  proj_kernel<<<128, 256, 0, stream>>>(ob, Wop, bo, x, x1);
  ln_kernel<<<NTOK, 128, 0, stream>>>(x1, g2, be2, h2b);
  ffn_kernel<<<FFN_SEG*32, 256, 0, stream>>>(h2b, W1p, W2p, bf1, part);
  ffn_reduce_kernel<<<(NTOK*E_DIM)/256, 256, 0, stream>>>(part, x1, bf2, out);

  (void)in_sizes; (void)n_in; (void)out_size; (void)ws_size;
}
